// KmerLSTM_CRF_82609400971563
// MI455X (gfx1250) — matrix operand order repaired and checked by reading, not compiled, not run on hardware
//
#include <hip/hip_runtime.h>
#include <hip/hip_bf16.h>
#include <math.h>

// ---------------- problem dims ----------------
#define TT   1024
#define BB   128
#define EE   128
#define HH   128
#define G4   512      // 4*H
#define CK   256      // E + H (fused gate GEMM K)
#define KTAG 64
#define XPAD 8        // bf16 elems of LDS row padding (16B = 4 dwords)

typedef __bf16 bf16;
typedef __attribute__((ext_vector_type(16))) __bf16 v16bf;
typedef __attribute__((ext_vector_type(8)))  __bf16 v8bf;
typedef __attribute__((ext_vector_type(8)))  float  v8f;
typedef __attribute__((ext_vector_type(4)))  float  v4f;
typedef unsigned int u32x4 __attribute__((ext_vector_type(4)));
typedef int          i32x4 __attribute__((ext_vector_type(4)));
typedef int          i32x8 __attribute__((ext_vector_type(8)));

union F16U { v16bf v; v8bf h[2]; };

// gfx1250 has a hardware v_tanh_f32 (TRANS32, co-executes with WMMA pipe)
#if defined(__has_builtin)
#  if __has_builtin(__builtin_amdgcn_tanhf)
#    define FAST_TANH(x) __builtin_amdgcn_tanhf(x)
#  endif
#endif
#ifndef FAST_TANH
#  define FAST_TANH(x) tanhf(x)
#endif

__device__ __forceinline__ float tanh_fast(float x) { return FAST_TANH(x); }
// sigmoid(x) = 0.5*tanh(x/2) + 0.5 : 1 TRANS op + FMA instead of exp+rcp chain
__device__ __forceinline__ float sigf(float x) { return __builtin_fmaf(0.5f, tanh_fast(0.5f * x), 0.5f); }

// ---------------- TDM: DMA one contiguous 4KB x-tile into (row-padded) LDS ----
// Tile = 16 rows x 128 bf16 = 1024 dwords, contiguous in global memory.
// LDS padding: insert 4 dwords after every 64 dwords -> 272B row stride
// (de-conflicts LDS banks for the transposed A-fragment ds_load_b128s).
__device__ __forceinline__ void tdm_load_x_tile(const bf16* gsrc, unsigned lds_off) {
    unsigned long long ga = (unsigned long long)(uintptr_t)gsrc;
    u32x4 g0;
    g0[0] = 1u;                                           // count=1 (valid user D#)
    g0[1] = lds_off;                                      // D#.lds_addr
    g0[2] = (unsigned)(ga & 0xFFFFFFFFull);               // global_addr[31:0]
    g0[3] = (unsigned)((ga >> 32) & 0x01FFFFFFull)        // global_addr[56:32]
          | (2u << 30);                                   // type=2 ("image")
    i32x8 g1;
    g1[0] = (2 << 16)        // data_size=2 (4-byte dwords), workgroup_mask=0
          | (1 << 20)        // pad_enable
          | (5 << 22)        // pad_interval: 2^(5+1)=64 dwords (256B) between pads
          | (3 << 25);       // pad_amount: 3+1=4 dwords (16B)
    g1[1] = (int)((1024u & 0xFFFFu) << 16);    // tensor_dim0[15:0]  (bits 63:48)
    g1[2] = (int)((1024u >> 16) | (1u << 16)); // tensor_dim0[31:16] | tensor_dim1=1
    g1[3] = (int)(1024u << 16);                // tile_dim0 = 1024 dwords
    g1[4] = 0;                                 // tile_dim1 = tile_dim2 = 0 (1-D tile)
    g1[5] = 1024;                              // tensor_dim0_stride[31:0]
    g1[6] = 0;
    g1[7] = 0;
    i32x4 z4 = {0, 0, 0, 0};
#if __clang_major__ >= 23
    i32x8 z8 = {0, 0, 0, 0, 0, 0, 0, 0};
    __builtin_amdgcn_tensor_load_to_lds(g0, g1, z4, z4, z8, 0);
#else
    __builtin_amdgcn_tensor_load_to_lds(g0, g1, z4, z4, 0);
#endif
}

// ---------------- kernel 1: embedding gather + bf16 cast (8 elems/thread) -----
__global__ __launch_bounds__(256) void embed_cast_kernel(
    const int* __restrict__ kmers, const float* __restrict__ emb,
    bf16* __restrict__ x)
{
    size_t c   = (size_t)blockIdx.x * 256 + threadIdx.x;  // chunk of 8 elems
    size_t row = c >> 4;                                  // E/8 = 16 chunks/row
    int    e0  = (int)(c & 15) * 8;
    int    kv  = kmers[row];
    const v4f* src = (const v4f*)(emb + (size_t)kv * EE + e0);
    v4f a = src[0], b = src[1];
    v8bf o;
    o[0] = (bf16)a[0]; o[1] = (bf16)a[1]; o[2] = (bf16)a[2]; o[3] = (bf16)a[3];
    o[4] = (bf16)b[0]; o[5] = (bf16)b[1]; o[6] = (bf16)b[2]; o[7] = (bf16)b[3];
    *(v8bf*)(x + row * EE + e0) = o;
}

// ---------------- kernel 2: weight prep ----------------
__global__ __launch_bounds__(256) void prep_weights_kernel(
    const float* __restrict__ wih_f, const float* __restrict__ whh_f,
    const float* __restrict__ bih_f, const float* __restrict__ bhh_f,
    const float* __restrict__ wih_b, const float* __restrict__ whh_b,
    const float* __restrict__ bih_b, const float* __restrict__ bhh_b,
    const float* __restrict__ wproj,
    bf16* __restrict__ Wf, bf16* __restrict__ Wb,
    float* __restrict__ bvf, float* __restrict__ bvb,
    bf16* __restrict__ wpb)
{
    int idx = blockIdx.x * 256 + threadIdx.x;
    if (idx < G4 * CK) {
        int j = idx >> 8, k = idx & 255;
        float vf = (k < EE) ? wih_f[j * EE + k] : whh_f[j * HH + (k - EE)];
        float vb = (k < EE) ? wih_b[j * EE + k] : whh_b[j * HH + (k - EE)];
        Wf[idx] = (bf16)vf;
        Wb[idx] = (bf16)vb;
    }
    if (idx < G4) {
        bvf[idx] = bih_f[idx] + bhh_f[idx];
        bvb[idx] = bih_b[idx] + bhh_b[idx];
    }
    if (idx < KTAG * CK) wpb[idx] = (bf16)wproj[idx];
}

// ---------------- kernel 3: fused bidirectional LSTM scan ----------------
// grid = 16 blocks: dir = blockIdx>>3, batch tile = (blockIdx&7)*16.
// Wave w owns columns {g*128 + w*16 + (lane&15) : g=0..3}: the i/f/g/o gate
// values of a given (row,unit) land in the same lane / same acc index, so the
// LSTM cell runs fully in registers (c state in 8 VGPRs/lane).
// x tiles are DMA'd by the Tensor Data Mover, double-buffered, row-padded.
__global__ __launch_bounds__(256) void lstm_scan_kernel(
    const bf16* __restrict__ x,      // [T][B][E]
    const bf16* __restrict__ Wf,     // [512][256]
    const bf16* __restrict__ Wb,
    const float* __restrict__ bvf,   // [512]
    const float* __restrict__ bvb,
    bf16* __restrict__ hout)         // [T][B][256]
{
    const int dir = blockIdx.x >> 3;
    const int b0  = (blockIdx.x & 7) * 16;
    const bf16*  W  = dir ? Wb : Wf;
    const float* bv = dir ? bvb : bvf;

    __shared__ bf16 xs[2][16][EE + XPAD];   // double-buffered TDM dest (~8.5KB)
    __shared__ bf16 hs[16][HH + XPAD];      // h_{t-1}                   (~4.3KB)

    const int tid  = threadIdx.x;
    const int lane = tid & 31;
    const int wave = tid >> 5;

    for (int i = tid; i < 16 * HH; i += 256) hs[i >> 7][i & 127] = (bf16)0.0f;

    const int uc = wave * 16 + (lane & 15);        // hidden unit owned by lane
    const float bI = bv[uc], bF = bv[128 + uc], bG = bv[256 + uc], bO = bv[384 + uc];
    float creg[8];
#pragma unroll
    for (int v = 0; v < 8; ++v) creg[v] = 0.0f;

    const int am  = lane & 15;                     // A-frag row
    const int akb = (lane >> 4) * 8;               // A-frag K sub-offset
    const int bkb = (lane >> 4) * 8;               // B-frag K sub-offset
    const bf16* wrow[4];
#pragma unroll
    for (int g = 0; g < 4; ++g)
        wrow[g] = W + (size_t)(g * 128 + uc) * CK + bkb;   // loop-invariant

    // DMA first x tile
    const int t0 = dir ? (TT - 1) : 0;
    if (wave == 0)
        tdm_load_x_tile(x + ((size_t)t0 * BB + b0) * EE,
                        (unsigned)(uintptr_t)&xs[0][0][0]);
    __syncthreads();    // publish hs init

    for (int step = 0; step < TT; ++step) {
        const int cur = step & 1;
        const int t   = dir ? (TT - 1 - step) : step;

        if (wave == 0) {
            if (step + 1 < TT) {
                const int tn = dir ? (TT - 2 - step) : (step + 1);
                tdm_load_x_tile(x + ((size_t)tn * BB + b0) * EE,
                                (unsigned)(uintptr_t)&xs[cur ^ 1][0][0]);
                __builtin_amdgcn_s_wait_tensorcnt((short)1);   // xs[cur] done
            } else {
                __builtin_amdgcn_s_wait_tensorcnt((short)0);
            }
        }
        __syncthreads();   // barrier A: xs[cur] ready, previous h published

        // A fragments: K 0..127 from x tile, K 128..255 from h
        v16bf afr[8];
#pragma unroll
        for (int kf = 0; kf < 4; ++kf) {
            F16U u;
            u.h[0] = *(const v8bf*)&xs[cur][am][kf * 32 + akb];
            u.h[1] = *(const v8bf*)&xs[cur][am][kf * 32 + akb + 16];
            afr[kf] = u.v;
        }
#pragma unroll
        for (int kf = 0; kf < 4; ++kf) {
            F16U u;
            u.h[0] = *(const v8bf*)&hs[am][kf * 32 + akb];
            u.h[1] = *(const v8bf*)&hs[am][kf * 32 + akb + 16];
            afr[4 + kf] = u.v;
        }

        v8f acc[4];
#pragma unroll
        for (int g = 0; g < 4; ++g) acc[g] = (v8f){};
#pragma unroll
        for (int kf = 0; kf < 8; ++kf) {
#pragma unroll
            for (int g = 0; g < 4; ++g) {
                F16U u;                               // loop-invariant -> hoisted
                u.h[0] = *(const v8bf*)(wrow[g] + kf * 32);
                u.h[1] = *(const v8bf*)(wrow[g] + kf * 32 + 16);
                acc[g] = __builtin_amdgcn_wmma_f32_16x16x32_bf16(
                            false, afr[kf], false, u.v, (short)0, acc[g], false, false);
            }
        }
        __syncthreads();   // barrier B: all hs reads complete before overwrite

        // in-register LSTM cell: acc[0..3][v] = i,f,g,o of (row mbase+v, unit uc)
        const int mbase = (lane >> 4) * 8;
        const size_t orow = ((size_t)t * BB + b0) * (size_t)(2 * HH)
                          + (size_t)(dir * HH + uc);
#pragma unroll
        for (int v = 0; v < 8; ++v) {
            float gi = acc[0][v] + bI;
            float gf = acc[1][v] + bF;
            float gg = acc[2][v] + bG;
            float go = acc[3][v] + bO;
            float c  = sigf(gf) * creg[v] + sigf(gi) * tanh_fast(gg);
            float h  = sigf(go) * tanh_fast(c);
            creg[v]  = c;
            bf16 hb  = (bf16)h;
            hs[mbase + v][uc] = hb;
            hout[orow + (size_t)(mbase + v) * (2 * HH)] = hb;
        }
    }
}

// ---------------- kernel 4: projection GEMM ----------------
// em[m][k] = out[m][:] . wproj[k][:] + bproj[k];  M = T*B, N = 64, K = 256
__global__ __launch_bounds__(256) void proj_wmma_kernel(
    const bf16* __restrict__ out,    // [T*B][256]
    const bf16* __restrict__ wp,     // [64][256]
    const float* __restrict__ bp,    // [64]
    float* __restrict__ em)          // [T*B][64]
{
    const int tid  = threadIdx.x;
    const int lane = tid & 31;
    const int wave = tid >> 5;
    const size_t m0 = ((size_t)blockIdx.x * 8 + wave) * 16;

    const bf16* arow = out + (m0 + (lane & 15)) * CK + (lane >> 4) * 8;

    v8f acc[4];
#pragma unroll
    for (int nt = 0; nt < 4; ++nt) acc[nt] = (v8f){};

#pragma unroll
    for (int kf = 0; kf < 8; ++kf) {
        F16U a;
        a.h[0] = *(const v8bf*)(arow + kf * 32);
        a.h[1] = *(const v8bf*)(arow + kf * 32 + 16);
#pragma unroll
        for (int nt = 0; nt < 4; ++nt) {
            const bf16* wrow = wp + (size_t)(nt * 16 + (lane & 15)) * CK + (lane >> 4) * 8;
            F16U b;
            b.h[0] = *(const v8bf*)(wrow + kf * 32);
            b.h[1] = *(const v8bf*)(wrow + kf * 32 + 16);
            acc[nt] = __builtin_amdgcn_wmma_f32_16x16x32_bf16(
                        false, a.v, false, b.v, (short)0, acc[nt], false, false);
        }
    }

    const int mg = (lane >> 4) * 8;
#pragma unroll
    for (int nt = 0; nt < 4; ++nt) {
        int n = nt * 16 + (lane & 15);
        float bb = bp[n];
#pragma unroll
        for (int v = 0; v < 8; ++v)
            em[(m0 + mg + v) * KTAG + n] = acc[nt][v] + bb;
    }
}

// ---------------- kernel 5: CRF (numerator + forward algorithm) ----------------
__global__ __launch_bounds__(64) void crf_forward_kernel(
    const float* __restrict__ em,       // [T][B][64]
    const int*   __restrict__ tags,     // [B]
    const float* __restrict__ start_t,  // [64]
    const float* __restrict__ end_t,    // [64]
    const float* __restrict__ trans,    // [64][64]
    float* __restrict__ partial)        // [B] : denominator - numerator
{
    __shared__ float tr[KTAG][KTAG];
    __shared__ float sc[KTAG];
    __shared__ float red[KTAG];
    __shared__ float emsum;

    const int b = blockIdx.x;
    const int j = threadIdx.x;

    for (int i = 0; i < KTAG; ++i) tr[i][j] = trans[i * KTAG + j];
    const int tag = tags[b];

    float es = 0.0f;
    for (int t = j; t < TT; t += KTAG)
        es += em[((size_t)t * BB + b) * KTAG + tag];
    red[j] = es;
    __syncthreads();
    for (int s = 32; s > 0; s >>= 1) { if (j < s) red[j] += red[j + s]; __syncthreads(); }
    if (j == 0) emsum = red[0];
    __syncthreads();

    sc[j] = start_t[j] + em[(size_t)b * KTAG + j];
    __syncthreads();
    for (int t = 1; t < TT; ++t) {
        float m = -3.0e38f;
        for (int i = 0; i < KTAG; ++i) m = fmaxf(m, sc[i] + tr[i][j]);
        float s = 0.0f;
        for (int i = 0; i < KTAG; ++i) s += __expf(sc[i] + tr[i][j] - m);
        float v = m + __logf(s) + em[((size_t)t * BB + b) * KTAG + j];
        __syncthreads();
        sc[j] = v;
        __syncthreads();
    }

    float fv = sc[j] + end_t[j];
    red[j] = fv;
    __syncthreads();
    for (int s = 32; s > 0; s >>= 1) { if (j < s) red[j] = fmaxf(red[j], red[j + s]); __syncthreads(); }
    float mx = red[0];
    __syncthreads();
    red[j] = __expf(fv - mx);
    __syncthreads();
    for (int s = 32; s > 0; s >>= 1) { if (j < s) red[j] += red[j + s]; __syncthreads(); }

    if (j == 0) {
        float denom = mx + __logf(red[0]);
        float numer = start_t[tag] + emsum + (float)(TT - 1) * tr[tag][tag] + end_t[tag];
        partial[b] = denom - numer;
    }
}

// ---------------- kernel 6: final reduction ----------------
__global__ __launch_bounds__(128) void final_reduce_kernel(
    const float* __restrict__ partial, float* __restrict__ out)
{
    __shared__ float r[BB];
    int t = threadIdx.x;
    r[t] = partial[t];
    __syncthreads();
    for (int s = 64; s > 0; s >>= 1) { if (t < s) r[t] += r[t + s]; __syncthreads(); }
    if (t == 0) out[0] = r[0];
}

// ---------------- workspace layout (bytes) ----------------
#define OFF_X     ((size_t)0)                    // bf16 [T*B*E]        32 MB
#define OFF_HOUT  ((size_t)33554432)             // bf16 [T*B*256]      64 MB
#define OFF_EM    ((size_t)100663296)            // f32  [T*B*64]       32 MB
#define OFF_WF    ((size_t)134217728)            // bf16 [512*256]
#define OFF_WB    ((size_t)134479872)
#define OFF_WP    ((size_t)134742016)            // bf16 [64*256]
#define OFF_BVF   ((size_t)134774784)            // f32  [512]
#define OFF_BVB   ((size_t)134776832)            // f32  [512]
#define OFF_PART  ((size_t)134778880)             // f32  [128]

extern "C" void kernel_launch(void* const* d_in, const int* in_sizes, int n_in,
                              void* d_out, int out_size, void* d_ws, size_t ws_size,
                              hipStream_t stream) {
    (void)in_sizes; (void)n_in; (void)out_size; (void)ws_size;

    const int*   kmers   = (const int*)  d_in[0];
    const int*   tags    = (const int*)  d_in[1];
    const float* emb     = (const float*)d_in[2];
    const float* wih_f   = (const float*)d_in[3];
    const float* whh_f   = (const float*)d_in[4];
    const float* bih_f   = (const float*)d_in[5];
    const float* bhh_f   = (const float*)d_in[6];
    const float* wih_b   = (const float*)d_in[7];
    const float* whh_b   = (const float*)d_in[8];
    const float* bih_b   = (const float*)d_in[9];
    const float* bhh_b   = (const float*)d_in[10];
    const float* wproj   = (const float*)d_in[11];
    const float* bproj   = (const float*)d_in[12];
    const float* start_t = (const float*)d_in[13];
    const float* end_t   = (const float*)d_in[14];
    const float* trans   = (const float*)d_in[15];

    char* ws = (char*)d_ws;
    bf16*  x    = (bf16*)(ws + OFF_X);
    bf16*  hout = (bf16*)(ws + OFF_HOUT);
    float* em   = (float*)(ws + OFF_EM);
    bf16*  Wf   = (bf16*)(ws + OFF_WF);
    bf16*  Wb   = (bf16*)(ws + OFF_WB);
    bf16*  wpb  = (bf16*)(ws + OFF_WP);
    float* bvf  = (float*)(ws + OFF_BVF);
    float* bvb  = (float*)(ws + OFF_BVB);
    float* part = (float*)(ws + OFF_PART);

    // 1. embedding gather + bf16 cast : T*B*E/8 threads = 8192 blocks
    embed_cast_kernel<<<8192, 256, 0, stream>>>(kmers, emb, x);

    // 2. weight prep
    prep_weights_kernel<<<512, 256, 0, stream>>>(
        wih_f, whh_f, bih_f, bhh_f, wih_b, whh_b, bih_b, bhh_b, wproj,
        Wf, Wb, bvf, bvb, wpb);

    // 3. bidirectional LSTM scan : 2 dirs x 8 batch tiles
    lstm_scan_kernel<<<16, 256, 0, stream>>>(x, Wf, Wb, bvf, bvb, hout);

    // 4. projection GEMM
    proj_wmma_kernel<<<1024, 256, 0, stream>>>(hout, wpb, bproj, em);

    // 5. CRF per batch element
    crf_forward_kernel<<<BB, 64, 0, stream>>>(em, tags, start_t, end_t, trans, part);

    // 6. scalar reduction
    final_reduce_kernel<<<1, 128, 0, stream>>>(part, (float*)d_out);
}
